// AdaptiveConv2DMod_1726576857281
// MI455X (gfx1250) — matrix operand order repaired and checked by reading, not compiled, not run on hardware
//
#include <hip/hip_runtime.h>

typedef __attribute__((ext_vector_type(16))) _Float16 v16h;
typedef __attribute__((ext_vector_type(8)))  _Float16 v8h;
typedef __attribute__((ext_vector_type(8)))  float    v8f;
typedef __attribute__((ext_vector_type(4)))  int      v4i;

#define BATCH 8
#define CI    32
#define CO    32
#define NK    4
#define HH    512
#define WW    512
#define KPOS  9
#define EPSD  1e-8f

// conv tile config
#define TY    4          // output rows per block
#define TX    128        // output cols per block
#define SROWS (TY + 2)   // input rows staged (halo)
#define SCOLS (TX + 2)   // input cols staged (halo)
#define PITCH 40         // f16 slots per staged pixel (80B stride: 16B-aligned, conflict-free cycling)
#define WPITCH 40        // f16 slots per weight output-channel row in LDS (same property)

// Async global->LDS copy support (gfx1250; guarded, with VGPR-path fallback)
#if defined(__AMDGCN__) && __has_builtin(__builtin_amdgcn_global_load_async_to_lds_b128)
#define USE_ASYNC_LDS 1
#else
#define USE_ASYNC_LDS 0
#endif

// ---------------------------------------------------------------------------
// Kernel 1: per-batch weight prep.
//   attn = softmax(kernel_mod[b]); wt = sum_n attn[n]*W[n]; wt *= (mod+1);
//   wt *= rsqrt(max(sum_{ic,k,k} wt^2, eps)); store f16 as [b][kpos][o][ic]
//   (A-fragment order: contiguous ic per output channel, 16B-aligned chunks).
// ---------------------------------------------------------------------------
__global__ void prep_weights_kernel(const float* __restrict__ mod,
                                    const float* __restrict__ kmod,
                                    const float* __restrict__ weights,
                                    _Float16* __restrict__ wq) {
    const int b   = blockIdx.x;
    const int tid = threadIdx.x;
    const int o   = tid >> 3;   // 0..31
    const int j   = tid & 7;    // 0..7

    __shared__ float wbuf[CO * CI * KPOS];   // [o][e], e = ic*9 + kpos  (36 KB)
    __shared__ float psum[256];
    __shared__ float inv_s[CO];

    float k0 = kmod[b * NK + 0], k1 = kmod[b * NK + 1];
    float k2 = kmod[b * NK + 2], k3 = kmod[b * NK + 3];
    float mx = fmaxf(fmaxf(k0, k1), fmaxf(k2, k3));
    float e0 = __expf(k0 - mx), e1 = __expf(k1 - mx);
    float e2 = __expf(k2 - mx), e3 = __expf(k3 - mx);
    float is = 1.0f / (e0 + e1 + e2 + e3);
    float a0 = e0 * is, a1 = e1 * is, a2 = e2 * is, a3 = e3 * is;

    const int NW = CO * CI * KPOS;  // 9216 elems per bank kernel
    float acc = 0.0f;
    for (int t = 0; t < 36; ++t) {
        int e  = j * 36 + t;              // 0..287 within this o
        int ic = e / 9;
        int kp = e - ic * 9;
        int gi = (o * CI + ic) * KPOS + kp;
        float w = a0 * weights[gi]
                + a1 * weights[gi + NW]
                + a2 * weights[gi + 2 * NW]
                + a3 * weights[gi + 3 * NW];
        w *= (mod[b * CI + ic] + 1.0f);
        wbuf[o * 288 + e] = w;
        acc += w * w;
    }
    psum[tid] = acc;
    __syncthreads();
    if (j == 0) {
        float s = 0.0f;
        for (int k = 0; k < 8; ++k) s += psum[o * 8 + k];
        inv_s[o] = rsqrtf(fmaxf(s, EPSD));
    }
    __syncthreads();
    const float inv = inv_s[o];
    for (int t = 0; t < 36; ++t) {
        int e  = j * 36 + t;
        int ic = e / 9;
        int kp = e - ic * 9;
        wq[((b * KPOS + kp) * CO + o) * CI + ic] =
            (_Float16)(wbuf[o * 288 + e] * inv);
    }
}

// ---------------------------------------------------------------------------
// Kernel 2: batch-grouped 3x3 conv as implicit GEMM on v_wmma_f32_16x16x32_f16.
//   Per block: one batch b, output tile 4 rows x 128 cols (512 px).
//   GEMM: out[32, 512] = A[32, 288] x B[288, 512], K = 9 kpos x 32 ic.
//   Weights (18 KB, already f16) copied via async global->LDS b128 (ASYNCcnt);
//   fmap halo staged via VGPR path with f32->f16 convert (needs transform).
// ---------------------------------------------------------------------------
__global__ void __launch_bounds__(256)
conv_wmma_kernel(const float* __restrict__ fmap,
                 const _Float16* __restrict__ wq,
                 float* __restrict__ out) {
    __shared__ alignas(16) _Float16 smem[SROWS * SCOLS * PITCH];   // 62400 B
    __shared__ alignas(16) _Float16 wlds[KPOS * CO * WPITCH];      // 23040 B

    const int b  = blockIdx.z;
    const int y0 = blockIdx.y * TY;
    const int x0 = blockIdx.x * TX;
    const int tid = threadIdx.x;

    // ---- stage per-batch weights (f16, raw bytes) into LDS with pitch-40 rows.
    // chunk i = 16B: global f16 offset i*8; LDS row = i>>2 (one (kp,o) pair), col = (i&3)*8.
    {
        const _Float16* __restrict__ gw = wq + b * KPOS * CO * CI;
        const int NCHUNK = KPOS * CO * 4;   // 1152 x 16B
        for (int i = tid; i < NCHUNK; i += 256) {
#if USE_ASYNC_LDS
            __builtin_amdgcn_global_load_async_to_lds_b128(
                (__attribute__((address_space(1))) v4i*)(gw + i * 8),
                (__attribute__((address_space(3))) v4i*)&wlds[(i >> 2) * WPITCH + (i & 3) * 8],
                0, 0);
#else
            v8h v = *(const v8h*)(gw + i * 8);
            *(v8h*)&wlds[(i >> 2) * WPITCH + (i & 3) * 8] = v;
#endif
        }
    }

    // ---- stage fmap halo tile: f32 global -> f16 LDS, transposed to [r][x][ic]
    // flat order (r, ic, x): consecutive threads hit consecutive x -> coalesced.
    const int TE = SROWS * CI * SCOLS;  // 24960
    for (int f = tid; f < TE; f += 256) {
        int r   = f / (CI * SCOLS);
        int rem = f - r * (CI * SCOLS);
        int ic  = rem / SCOLS;
        int x   = rem - ic * SCOLS;
        int gy  = y0 - 1 + r;
        int gx  = x0 - 1 + x;
        float v = 0.0f;
        if ((unsigned)gy < (unsigned)HH && (unsigned)gx < (unsigned)WW)
            v = fmap[((b * CI + ic) * HH + gy) * WW + gx];
        smem[(r * SCOLS + x) * PITCH + ic] = (_Float16)v;
    }

#if USE_ASYNC_LDS
#if __has_builtin(__builtin_amdgcn_s_wait_asynccnt)
    __builtin_amdgcn_s_wait_asynccnt(0);
#else
    asm volatile("s_wait_asynccnt 0x0" ::: "memory");
#endif
#endif
    __syncthreads();

    const int w    = tid >> 5;   // wave 0..7
    const int lane = tid & 31;
    const int half = lane >> 4;  // 0: lanes 0-15, 1: lanes 16-31
    const int lm   = lane & 15;

    v8f acc[4][2];
#pragma unroll
    for (int nt = 0; nt < 4; ++nt)
#pragma unroll
        for (int m = 0; m < 2; ++m)
            acc[nt][m] = (v8f){0.f, 0.f, 0.f, 0.f, 0.f, 0.f, 0.f, 0.f};

#pragma unroll
    for (int kp = 0; kp < KPOS; ++kp) {
        const int ky = kp / 3;
        const int kx = kp - 3 * ky;

        // A fragments from LDS: 16x32 f16 layout (lane<16: K=ic 0-7 then 16-23;
        // lane>=16: K=ic 8-15 then 24-31), rows = output channels.
        const _Float16* __restrict__ wk = &wlds[kp * CO * WPITCH];
        v16h afrag[2];
#pragma unroll
        for (int m = 0; m < 2; ++m) {
            const int o = m * 16 + lm;
            v8h lo = *(const v8h*)(wk + o * WPITCH + half * 8);       // ic 0-7 / 8-15
            v8h hi = *(const v8h*)(wk + o * WPITCH + 16 + half * 8);  // ic 16-23 / 24-31
#pragma unroll
            for (int i = 0; i < 8; ++i) { afrag[m][i] = lo[i]; afrag[m][8 + i] = hi[i]; }
        }

#pragma unroll
        for (int nt = 0; nt < 4; ++nt) {
            const int p  = w * 64 + nt * 16 + lm;   // pixel id in tile, col N = lm
            const int py = p >> 7;
            const int px = p & (TX - 1);
            // B fragment: 32x16 f16 layout (lane<16: K=ic 0-7 then 16-23; lane>=16: ic 8-15 then 24-31)
            const _Float16* __restrict__ sp =
                &smem[((py + ky) * SCOLS + (px + kx)) * PITCH + half * 8];
            v8h b0 = *(const v8h*)sp;
            v8h b1 = *(const v8h*)(sp + 16);
            v16h bfrag;
#pragma unroll
            for (int i = 0; i < 8; ++i) { bfrag[i] = b0[i]; bfrag[8 + i] = b1[i]; }

            acc[nt][0] = __builtin_amdgcn_wmma_f32_16x16x32_f16(
                false, afrag[0], false, bfrag, (short)0, acc[nt][0], false, false);
            acc[nt][1] = __builtin_amdgcn_wmma_f32_16x16x32_f16(
                false, afrag[1], false, bfrag, (short)0, acc[nt][1], false, false);
        }
    }

    // ---- store: C/D layout: VGPR r -> row M = r (lanes 0-15) or r+8 (lanes 16-31)
#pragma unroll
    for (int nt = 0; nt < 4; ++nt) {
        const int p = w * 64 + nt * 16 + lm;
        const int y = y0 + (p >> 7);
        const int x = x0 + (p & (TX - 1));
#pragma unroll
        for (int m = 0; m < 2; ++m) {
#pragma unroll
            for (int r = 0; r < 8; ++r) {
                const int o = m * 16 + half * 8 + r;
                out[((b * CO + o) * HH + y) * WW + x] = acc[nt][m][r];
            }
        }
    }
}

// ---------------------------------------------------------------------------
extern "C" void kernel_launch(void* const* d_in, const int* in_sizes, int n_in,
                              void* d_out, int out_size, void* d_ws, size_t ws_size,
                              hipStream_t stream) {
    const float* fmap = (const float*)d_in[0];   // [8,32,512,512]
    const float* mod  = (const float*)d_in[1];   // [8,32]
    const float* kmod = (const float*)d_in[2];   // [8,4]
    const float* wts  = (const float*)d_in[3];   // [4,32,32,3,3]
    float* out = (float*)d_out;                  // [8,32,512,512]

    _Float16* wq = (_Float16*)d_ws;              // [8][9][32][32] f16 = 147456 B

    prep_weights_kernel<<<dim3(BATCH), dim3(256), 0, stream>>>(mod, kmod, wts, wq);
    conv_wmma_kernel<<<dim3(WW / TX, HH / TY, BATCH), dim3(256), 0, stream>>>(fmap, wq, out);
}
